// CrossAttentionModule_326417515340
// MI455X (gfx1250) — matrix operand order repaired and checked by reading, not compiled, not run on hardware
//
#include <hip/hip_runtime.h>
#include <hip/hip_bf16.h>

// ---------------------------------------------------------------------------
// Transformer encoder forward for MI455X (gfx1250, wave32, WMMA + async-LDS).
// DIM=1024, DEPTH=4, HEADS=16, DIM_HEAD=64, MLP=4096, B=2, N=1024.
// ---------------------------------------------------------------------------

typedef __attribute__((ext_vector_type(16))) _Float16 v16h;
typedef __attribute__((ext_vector_type(8)))  _Float16 v8h;
typedef __attribute__((ext_vector_type(4)))  _Float16 v4h;
typedef __attribute__((ext_vector_type(2)))  _Float16 v2h;
typedef __attribute__((ext_vector_type(8)))  float    v8f;

#define DIM      1024
#define HEADS    16
#define DIM_HEAD 64
#define INNER    1024
#define MLP      4096
#define BATCH    2
#define SEQ      1024
#define NTOK     (BATCH * SEQ)   // 2048

// padded LDS row strides (in halves): 40 -> 80B = 20 banks, 72 -> 144B = 36
// banks; gcd(20,64)=gcd(36,64)=4 with period 16, so 16 consecutive rows map
// to 16 distinct bank groups -> conflict-free 16-lane fragment loads.
#define PAD32    40
#define PAD64    72

// ---- fragment helpers (A and B share contiguous k order, not ISA table order) ----------------------
// A fragment (16x32 f16): lane row = lane%16; halves e hold K=16*(lane/16)+e
//  -> one contiguous 32B chunk from a [M][K] tile.
// B fragment (32x16 f16): lane col = lane%16; halves e hold K=16*(lane/16)+e
//  -> one contiguous 32B chunk from a [N][K] (transposed) tile.
// C/D fragment (16x16 f32): element r -> row r + 8*(lane/16), col lane%16.

__device__ __forceinline__ v16h ld_frag_split(const _Float16* lo, const _Float16* hi) {
  v8h a = *(const v8h*)lo;
  v8h b = *(const v8h*)hi;
  v16h r;
#pragma unroll
  for (int i = 0; i < 8; ++i) { r[i] = a[i]; r[i + 8] = b[i]; }
  return r;
}
__device__ __forceinline__ v16h ld_frag16(const _Float16* p) {
  return ld_frag_split(p, p + 8);
}
__device__ __forceinline__ v8f wmma_f16(v16h a, v16h b, v8f c) {
  return __builtin_amdgcn_wmma_f32_16x16x32_f16(false, a, false, b, (short)0, c,
                                                false, false);
}

// ---- CDNA5 async global->LDS copy (ASYNCcnt-tracked, bypasses VGPRs) -------
__device__ __forceinline__ void async_copy_b128(void* lds, const void* gptr) {
  unsigned l = (unsigned)(uintptr_t)lds;  // low 32 bits of generic = LDS offset
  asm volatile("global_load_async_to_lds_b128 %0, %1, off"
               :: "v"(l), "v"(gptr) : "memory");
}
__device__ __forceinline__ void wait_async0() {
  asm volatile("s_wait_asynccnt 0x0" ::: "memory");
}

// ---------------------------------------------------------------------------
// LayerNorm: one block per row (NTOK rows), 256 threads, 4 elems/thread.
// ---------------------------------------------------------------------------
__global__ __launch_bounds__(256)
void ln_kernel(const float* __restrict__ x, const float* __restrict__ g,
               const float* __restrict__ bta, _Float16* __restrict__ out) {
  __shared__ float red[8];
  const int row  = blockIdx.x;
  const int tid  = threadIdx.x;
  const int wave = tid >> 5;
  const int lane = tid & 31;
  const float* xr = x + (size_t)row * DIM;
  float v[4];
  float s = 0.0f;
#pragma unroll
  for (int i = 0; i < 4; ++i) { v[i] = xr[tid + 256 * i]; s += v[i]; }
#pragma unroll
  for (int off = 1; off < 32; off <<= 1) s += __shfl_xor(s, off);
  if (lane == 0) red[wave] = s;
  __syncthreads();
  float tot = 0.0f;
#pragma unroll
  for (int w = 0; w < 8; ++w) tot += red[w];
  const float mu = tot * (1.0f / DIM);
  float vs = 0.0f;
#pragma unroll
  for (int i = 0; i < 4; ++i) { float d = v[i] - mu; vs += d * d; }
  __syncthreads();
#pragma unroll
  for (int off = 1; off < 32; off <<= 1) vs += __shfl_xor(vs, off);
  if (lane == 0) red[wave] = vs;
  __syncthreads();
  float var = 0.0f;
#pragma unroll
  for (int w = 0; w < 8; ++w) var += red[w];
  var *= (1.0f / DIM);
  const float rstd = rsqrtf(var + 1e-5f);
#pragma unroll
  for (int i = 0; i < 4; ++i) {
    int c = tid + 256 * i;
    out[(size_t)row * DIM + c] = (_Float16)((v[i] - mu) * rstd * g[c] + bta[c]);
  }
}

// ---------------------------------------------------------------------------
// Tiled WMMA GEMM:  out = A(f16 [M][K]) * W(f32 [K][N]) (+bias)(+res)(gelu?)
// Block tile 128x128, 8 waves -> each wave 64x32 (4x2 C fragments), K-step 32.
// A tile is DMA'd with async global->LDS b128; overlapping it, W is read f32
// (L2-resident: 192MB), converted to f16 and transposed into LDS.
// ---------------------------------------------------------------------------
template <bool HAS_BIAS, bool HAS_RES, bool DO_GELU, bool OUT_F32, bool OUT_F16>
__global__ __launch_bounds__(256)
void gemm_kernel(const _Float16* __restrict__ A, const float* __restrict__ W,
                 const float* __restrict__ bias, const float* res,
                 float* outf, _Float16* __restrict__ outh,
                 int M, int K, int N) {
  __shared__ __align__(16) _Float16 As[128][PAD32];
  __shared__ __align__(16) _Float16 Bs[128][PAD32];  // Bs[n][k] = W[k0+k][n0+n]
  const int tid  = threadIdx.x;
  const int wave = tid >> 5;
  const int lane = tid & 31;
  const int lhi  = lane >> 4;
  const int llo  = lane & 15;
  const int wm   = wave >> 2;  // 0..1
  const int wn   = wave & 3;   // 0..3
  const int m0   = blockIdx.y * 128;
  const int n0   = blockIdx.x * 128;

  v8f acc[4][2] = {};

  for (int k0 = 0; k0 < K; k0 += 32) {
    __syncthreads();
    // A tile: 128 rows x 32 halves, 512 async b128 transfers (2/thread)
#pragma unroll
    for (int c = 0; c < 2; ++c) {
      int chunk = tid + 256 * c;
      int row = chunk >> 2;
      int seg = chunk & 3;
      async_copy_b128(&As[row][seg * 8],
                      &A[(size_t)(m0 + row) * K + (k0 + seg * 8)]);
    }
    // W tile: 32x128 f32 -> f16, transposed; packed dword LDS stores
#pragma unroll
    for (int i = 0; i < 8; ++i) {
      int idx = tid + 256 * i;        // 2048 (n, k-pair) cells
      int nn  = idx & 127;
      int kp  = idx >> 7;             // 0..15 -> k = 2*kp
      _Float16 h0 = (_Float16)W[(size_t)(k0 + 2 * kp) * N + (n0 + nn)];
      _Float16 h1 = (_Float16)W[(size_t)(k0 + 2 * kp + 1) * N + (n0 + nn)];
      v2h pk; pk[0] = h0; pk[1] = h1;
      *(v2h*)&Bs[nn][2 * kp] = pk;
    }
    if (k0 + 32 < K)  // hint next weight tile toward L2 (global_prefetch_b8)
      __builtin_prefetch(&W[(size_t)(k0 + 32) * N + n0 + (tid & 127)], 0, 1);
    wait_async0();
    __syncthreads();

    v16h af[4];
#pragma unroll
    for (int mt = 0; mt < 4; ++mt) {
      const _Float16* p = &As[wm * 64 + mt * 16 + llo][lhi * 16];
      af[mt] = ld_frag16(p);
    }
    v16h bf[2];
#pragma unroll
    for (int nt = 0; nt < 2; ++nt)
      bf[nt] = ld_frag16(&Bs[wn * 32 + nt * 16 + llo][lhi * 16]);

#pragma unroll
    for (int mt = 0; mt < 4; ++mt)
#pragma unroll
      for (int nt = 0; nt < 2; ++nt)
        acc[mt][nt] = wmma_f16(af[mt], bf[nt], acc[mt][nt]);
  }

#pragma unroll
  for (int mt = 0; mt < 4; ++mt) {
#pragma unroll
    for (int nt = 0; nt < 2; ++nt) {
      int n = n0 + wn * 32 + nt * 16 + llo;
      float bv = HAS_BIAS ? bias[n] : 0.0f;
#pragma unroll
      for (int r = 0; r < 8; ++r) {
        int m = m0 + wm * 64 + mt * 16 + r + 8 * lhi;
        float v = acc[mt][nt][r] + bv;
        if (HAS_RES) v += res[(size_t)m * N + n];
        if (DO_GELU) v = 0.5f * v * (1.0f + erff(v * 0.70710678118654752f));
        if (OUT_F32) outf[(size_t)m * N + n] = v;
        if (OUT_F16) outh[(size_t)m * N + n] = (_Float16)v;
      }
    }
  }
}

// ---------------------------------------------------------------------------
// Fused flash attention: one block = (batch b, head h, 128 q rows).
// 8 waves x 16 q rows; keys in steps of 32 with online softmax; K tile is
// async-DMA'd to LDS, V is transposed via packed dword stores. Score tensor
// (128MB) never touches HBM.
// ---------------------------------------------------------------------------
__global__ __launch_bounds__(256)
void attn_kernel(const _Float16* __restrict__ qkv, const int* __restrict__ mask,
                 _Float16* __restrict__ o) {
  __shared__ __align__(16) _Float16 Ks[32][PAD64];       // [key][d]
  __shared__ __align__(16) _Float16 VTs[64][PAD32];      // [d][key]
  __shared__ __align__(16) _Float16 Ps[8][16][PAD32];    // per-wave P staging

  const int tid  = threadIdx.x;
  const int wave = tid >> 5;
  const int lane = tid & 31;
  const int lhi  = lane >> 4;
  const int llo  = lane & 15;

  const int qb = blockIdx.x;             // 0..7
  const int h  = blockIdx.y;             // 0..15
  const int b  = blockIdx.z;             // 0..1
  const int qrow = qb * 128 + wave * 16; // in-batch q row base for this wave

  // Q fragments, reused across all key steps (K-dim = 64 -> 2 fragments)
  v16h qa[2];
  {
    const _Float16* base =
        qkv + (size_t)(b * SEQ + qrow + llo) * (3 * INNER) + h * DIM_HEAD;
#pragma unroll
    for (int j = 0; j < 2; ++j) {
      const _Float16* p = base + 32 * j + 16 * lhi;
      qa[j] = ld_frag16(p);
    }
  }

  float mrow[8], lrow[8];
#pragma unroll
  for (int r = 0; r < 8; ++r) { mrow[r] = -1e30f; lrow[r] = 0.0f; }
  v8f oacc[4] = {};
  const float scale = 0.125f;  // 64^-0.5

  for (int kb = 0; kb < SEQ; kb += 32) {
    __syncthreads();
    {  // K tile: async DMA (32 keys x 64 dims, 1 b128/thread)
      int key = tid >> 3, seg = tid & 7;
      async_copy_b128(&Ks[key][seg * 8],
                      &qkv[(size_t)(b * SEQ + kb + key) * (3 * INNER) + INNER +
                           h * DIM_HEAD + seg * 8]);
    }
    {  // V^T tile: 2 keys x 4 dims per thread, packed dword stores
      int p2 = tid >> 4;   // key pair 0..15 -> keys 2*p2, 2*p2+1
      int dg = tid & 15;   // dim group -> dims dg*4 .. dg*4+3
      size_t r0 = (size_t)(b * SEQ + kb + 2 * p2) * (3 * INNER) + 2 * INNER +
                  h * DIM_HEAD + dg * 4;
      v4h a0 = *(const v4h*)&qkv[r0];
      v4h a1 = *(const v4h*)&qkv[r0 + 3 * INNER];
#pragma unroll
      for (int i = 0; i < 4; ++i) {
        v2h pk; pk[0] = a0[i]; pk[1] = a1[i];
        *(v2h*)&VTs[dg * 4 + i][2 * p2] = pk;
      }
    }
    wait_async0();
    __syncthreads();

    // S = Q * K^T for 16 q rows x 32 keys (two 16x16 C fragments)
    v8f s0 = {}, s1 = {};
#pragma unroll
    for (int j = 0; j < 2; ++j) {
      v16h kf0 = ld_frag16(&Ks[llo][32 * j + 16 * lhi]);
      v16h kf1 = ld_frag16(&Ks[16 + llo][32 * j + 16 * lhi]);
      s0 = wmma_f16(qa[j], kf0, s0);
      s1 = wmma_f16(qa[j], kf1, s1);
    }

    // mask + scale, row-max over 32 keys (16-lane xor reduction)
    float p0[8], p1[8], mx[8];
#pragma unroll
    for (int r = 0; r < 8; ++r) {
      int qr = qrow + r + 8 * lhi;
      const int* mp = mask + ((size_t)b * SEQ + qr) * SEQ + kb;
      float v0 = mp[llo]      ? s0[r] * scale : -1e9f;
      float v1 = mp[llo + 16] ? s1[r] * scale : -1e9f;
      p0[r] = v0; p1[r] = v1;
      mx[r] = fmaxf(v0, v1);
    }
#pragma unroll
    for (int off = 1; off < 16; off <<= 1)
#pragma unroll
      for (int r = 0; r < 8; ++r) mx[r] = fmaxf(mx[r], __shfl_xor(mx[r], off));

    float sum[8];
#pragma unroll
    for (int r = 0; r < 8; ++r) {
      float mn   = fmaxf(mrow[r], mx[r]);
      float corr = __expf(mrow[r] - mn);
      p0[r] = __expf(p0[r] - mn);
      p1[r] = __expf(p1[r] - mn);
      mrow[r] = mn;
      sum[r]  = p0[r] + p1[r];
      lrow[r] *= corr;
#pragma unroll
      for (int nt = 0; nt < 4; ++nt) oacc[nt][r] = oacc[nt][r] * corr;
    }
#pragma unroll
    for (int off = 1; off < 16; off <<= 1)
#pragma unroll
      for (int r = 0; r < 8; ++r) sum[r] += __shfl_xor(sum[r], off);
#pragma unroll
    for (int r = 0; r < 8; ++r) lrow[r] += sum[r];

    // C-layout P -> LDS -> A-layout fragment (wave-private, in-order DS)
#pragma unroll
    for (int r = 0; r < 8; ++r) {
      int row = r + 8 * lhi;
      Ps[wave][row][llo]      = (_Float16)p0[r];
      Ps[wave][row][llo + 16] = (_Float16)p1[r];
    }
    asm volatile("s_wait_dscnt 0x0" ::: "memory");
    const _Float16* pp = &Ps[wave][llo][16 * lhi];
    v16h pa = ld_frag16(pp);
#pragma unroll
    for (int nt = 0; nt < 4; ++nt) {
      v16h vf = ld_frag16(&VTs[nt * 16 + llo][16 * lhi]);
      oacc[nt] = wmma_f16(pa, vf, oacc[nt]);
    }
  }

  // normalize and write O (f16, [token][h*64+d]) for the output projection
#pragma unroll
  for (int nt = 0; nt < 4; ++nt)
#pragma unroll
    for (int r = 0; r < 8; ++r) {
      int t = b * SEQ + qrow + r + 8 * lhi;
      o[(size_t)t * INNER + h * DIM_HEAD + nt * 16 + llo] =
          (_Float16)(oacc[nt][r] / lrow[r]);
    }
}

// ---------------------------------------------------------------------------
// Host-side orchestration
// ---------------------------------------------------------------------------
extern "C" void kernel_launch(void* const* d_in, const int* in_sizes, int n_in,
                              void* d_out, int out_size, void* d_ws,
                              size_t ws_size, hipStream_t stream) {
  (void)in_sizes; (void)n_in; (void)out_size; (void)ws_size;
  const float* x_in  = (const float*)d_in[0];
  const int*   mask  = (const int*)d_in[1];
  const float* ln1_g = (const float*)d_in[2];
  const float* ln1_b = (const float*)d_in[3];
  const float* qkv_w = (const float*)d_in[4];
  const float* out_w = (const float*)d_in[5];
  const float* out_b = (const float*)d_in[6];
  const float* ln2_g = (const float*)d_in[7];
  const float* ln2_b = (const float*)d_in[8];
  const float* ff1_w = (const float*)d_in[9];
  const float* ff1_b = (const float*)d_in[10];
  const float* ff2_w = (const float*)d_in[11];
  const float* ff2_b = (const float*)d_in[12];

  // workspace carve-out (all 16B aligned): total ~44MB
  char* w = (char*)d_ws;
  float*     x    = (float*)w;     w += (size_t)NTOK * DIM * 4;       // 8MB
  _Float16*  hbuf = (_Float16*)w;  w += (size_t)NTOK * DIM * 2;       // 4MB
  _Float16*  qkvb = (_Float16*)w;  w += (size_t)NTOK * 3 * INNER * 2; // 12MB
  _Float16*  obuf = (_Float16*)w;  w += (size_t)NTOK * INNER * 2;     // 4MB
  _Float16*  mlpb = (_Float16*)w;  w += (size_t)NTOK * MLP * 2;       // 16MB

  hipMemcpyAsync(x, x_in, (size_t)NTOK * DIM * 4, hipMemcpyDeviceToDevice,
                 stream);

  const dim3 blk(256);
  for (int l = 0; l < 4; ++l) {
    // --- attention block ---
    ln_kernel<<<NTOK, blk, 0, stream>>>(x, ln1_g + l * DIM, ln1_b + l * DIM,
                                        hbuf);
    gemm_kernel<false, false, false, false, true>
        <<<dim3(3 * INNER / 128, NTOK / 128), blk, 0, stream>>>(
            hbuf, qkv_w + (size_t)l * DIM * 3 * INNER, nullptr, nullptr,
            nullptr, qkvb, NTOK, DIM, 3 * INNER);
    attn_kernel<<<dim3(SEQ / 128, HEADS, BATCH), blk, 0, stream>>>(qkvb, mask,
                                                                   obuf);
    gemm_kernel<true, true, false, true, false>
        <<<dim3(DIM / 128, NTOK / 128), blk, 0, stream>>>(
            obuf, out_w + (size_t)l * INNER * DIM, out_b + (size_t)l * DIM, x,
            x, nullptr, NTOK, INNER, DIM);
    // --- feedforward block ---
    ln_kernel<<<NTOK, blk, 0, stream>>>(x, ln2_g + l * DIM, ln2_b + l * DIM,
                                        hbuf);
    gemm_kernel<true, false, true, false, true>
        <<<dim3(MLP / 128, NTOK / 128), blk, 0, stream>>>(
            hbuf, ff1_w + (size_t)l * DIM * MLP, ff1_b + (size_t)l * MLP,
            nullptr, nullptr, mlpb, NTOK, DIM, MLP);
    gemm_kernel<true, true, false, true, false>
        <<<dim3(DIM / 128, NTOK / 128), blk, 0, stream>>>(
            mlpb, ff2_w + (size_t)l * MLP * DIM, ff2_b + (size_t)l * DIM, x, x,
            nullptr, NTOK, MLP, DIM);
  }

  hipMemcpyAsync(d_out, x, (size_t)NTOK * DIM * 4, hipMemcpyDeviceToDevice,
                 stream);
}
